// LightweightTransformerMoE_66116726555016
// MI455X (gfx1250) — compile-verified, matrix operand order checked by reading, not run on hardware
//
#include <hip/hip_runtime.h>

// ---------------- model dims ----------------
#define NTOK   8192      // B*S
#define DMODEL 256
#define SEQ    512
#define BATCH  16
#define NHEAD  8
#define DHEAD  32
#define NEXP   8
#define EHID   1024
#define PERMCAP 8448     // 8192 + 8*32 padding slots (32-aligned expert segments)

typedef __attribute__((ext_vector_type(16))) _Float16 v16h;
typedef __attribute__((ext_vector_type(8)))  _Float16 v8h;
typedef __attribute__((ext_vector_type(8)))  float    v8f;
typedef __attribute__((ext_vector_type(4)))  float    v4f;

union HV { v16h v; v8h h[2]; _Float16 e[16]; };
union CV { v8f  v; float f[8]; };

static __device__ __forceinline__ int imin(int a, int b) { return a < b ? a : b; }

static __device__ __forceinline__ v8f wmma_f16(v16h a, v16h b, v8f c) {
  // D = A(16x32 f16) * B(32x16 f16) + C(16x16 f32)
  return __builtin_amdgcn_wmma_f32_16x16x32_f16(false, a, false, b, (short)0, c,
                                                false, false);
}

// chunk loaders: 8 contiguous K elements for the lane (ISA 16-bit A/B layout)
static __device__ __forceinline__ v8h loadA8(const _Float16* p) {
  return *(const v8h*)p;
}
static __device__ __forceinline__ v8h loadA8(const float* p) {
  v4f a = *(const v4f*)p;
  v4f b = *(const v4f*)(p + 4);
  v8h r;
  r[0] = (_Float16)a[0]; r[1] = (_Float16)a[1];
  r[2] = (_Float16)a[2]; r[3] = (_Float16)a[3];
  r[4] = (_Float16)b[0]; r[5] = (_Float16)b[1];
  r[6] = (_Float16)b[2]; r[7] = (_Float16)b[3];
  return r;
}

// ---------------- WMMA GEMM, 32x64 tile per wave: C = A[M,K] @ W[N,K]^T + bias ----------
// W is f16 row-major [N,K] (== WMMA B column layout with contiguous K chunks)
template <typename TA, bool RELU, bool HALF_OUT>
__global__ void wmma_gemm(const TA* __restrict__ A, const _Float16* __restrict__ Bw,
                          const float* __restrict__ bias, void* __restrict__ Cout,
                          int M, int Nn, int K) {
  const int lane = threadIdx.x;
  const int g = lane >> 4, mr = lane & 15;
  const int tm = blockIdx.x, tn = blockIdx.y;

  const TA* Ar[2];
#pragma unroll
  for (int mi = 0; mi < 2; ++mi)
    Ar[mi] = A + (size_t)imin(tm * 32 + mi * 16 + mr, M - 1) * K;
  const _Float16* Br[4];
#pragma unroll
  for (int s = 0; s < 4; ++s)
    Br[s] = Bw + (size_t)imin(tn * 64 + s * 16 + mr, Nn - 1) * K;

  HV a[2], bb[4];
  CV c[2][4];
#pragma unroll
  for (int mi = 0; mi < 2; ++mi)
#pragma unroll
    for (int s = 0; s < 4; ++s)
#pragma unroll
      for (int i = 0; i < 8; ++i) c[mi][s].f[i] = 0.0f;

  for (int kk = 0; kk < K; kk += 32) {
    const int k0 = kk + 8 * g;
    const int k1 = kk + 16 + 8 * g;
#pragma unroll
    for (int mi = 0; mi < 2; ++mi) {
      a[mi].h[0] = loadA8(Ar[mi] + k0);
      a[mi].h[1] = loadA8(Ar[mi] + k1);
    }
#pragma unroll
    for (int s = 0; s < 4; ++s) {
      __builtin_prefetch(Br[s] + k0 + 64, 0, 1);
      bb[s].h[0] = *(const v8h*)(Br[s] + k0);
      bb[s].h[1] = *(const v8h*)(Br[s] + k1);
    }
#pragma unroll
    for (int mi = 0; mi < 2; ++mi)
#pragma unroll
      for (int s = 0; s < 4; ++s)
        c[mi][s].v = wmma_f16(a[mi].v, bb[s].v, c[mi][s].v);
  }

#pragma unroll
  for (int s = 0; s < 4; ++s) {
    const int col = tn * 64 + s * 16 + mr;
    if (col >= Nn) continue;
    const float bv = bias ? bias[col] : 0.0f;
#pragma unroll
    for (int mi = 0; mi < 2; ++mi) {
#pragma unroll
      for (int j = 0; j < 8; ++j) {
        const int row = tm * 32 + mi * 16 + j + 8 * g;
        if (row < M) {
          float x = c[mi][s].f[j] + bv;
          if (RELU) x = fmaxf(x, 0.0f);
          if (HALF_OUT)
            ((_Float16*)Cout)[(size_t)row * Nn + col] = (_Float16)x;
          else
            ((float*)Cout)[(size_t)row * Nn + col] = x;
        }
      }
    }
  }
}

// ---------------- MoE routed GEMM (rows grouped by expert, 32-aligned segments) ---------
template <typename TA, bool RELU, bool HALF_OUT>
__global__ void wmma_gemm_moe(const TA* __restrict__ A, const _Float16* __restrict__ BwBase,
                              const float* __restrict__ biasBase, void* __restrict__ Cout,
                              const int* __restrict__ eoff, const int* __restrict__ rowIdx,
                              int Nn, int K) {
  const int row0 = blockIdx.x * 32;
  const int total = eoff[NEXP];
  if (row0 >= total) return;                  // uniform across the wave
  int e = 0;
  while (e < NEXP - 1 && row0 >= eoff[e + 1]) ++e;

  const int lane = threadIdx.x;
  const int g = lane >> 4, mr = lane & 15;
  const int tn = blockIdx.y;

  const TA* Ar[2];
#pragma unroll
  for (int mi = 0; mi < 2; ++mi) {
    const int pr = imin(row0 + mi * 16 + mr, total - 1);
    const int arow = rowIdx ? rowIdx[pr] : pr;
    Ar[mi] = A + (size_t)arow * K;
  }
  const _Float16* Bexp = BwBase + (size_t)e * Nn * K;
  const _Float16* Br[4];
#pragma unroll
  for (int s = 0; s < 4; ++s)
    Br[s] = Bexp + (size_t)imin(tn * 64 + s * 16 + mr, Nn - 1) * K;
  const float* bias = biasBase + (size_t)e * Nn;

  HV a[2], bb[4];
  CV c[2][4];
#pragma unroll
  for (int mi = 0; mi < 2; ++mi)
#pragma unroll
    for (int s = 0; s < 4; ++s)
#pragma unroll
      for (int i = 0; i < 8; ++i) c[mi][s].f[i] = 0.0f;

  for (int kk = 0; kk < K; kk += 32) {
    const int k0 = kk + 8 * g;
    const int k1 = kk + 16 + 8 * g;
#pragma unroll
    for (int mi = 0; mi < 2; ++mi) {
      a[mi].h[0] = loadA8(Ar[mi] + k0);
      a[mi].h[1] = loadA8(Ar[mi] + k1);
    }
#pragma unroll
    for (int s = 0; s < 4; ++s) {
      __builtin_prefetch(Br[s] + k0 + 64, 0, 1);
      bb[s].h[0] = *(const v8h*)(Br[s] + k0);
      bb[s].h[1] = *(const v8h*)(Br[s] + k1);
    }
#pragma unroll
    for (int mi = 0; mi < 2; ++mi)
#pragma unroll
      for (int s = 0; s < 4; ++s)
        c[mi][s].v = wmma_f16(a[mi].v, bb[s].v, c[mi][s].v);
  }

#pragma unroll
  for (int s = 0; s < 4; ++s) {
    const int col = tn * 64 + s * 16 + mr;
    if (col >= Nn) continue;
    const float bv = bias[col];
#pragma unroll
    for (int mi = 0; mi < 2; ++mi) {
#pragma unroll
      for (int j = 0; j < 8; ++j) {
        const int row = row0 + mi * 16 + j + 8 * g;
        if (row < total) {
          float x = c[mi][s].f[j] + bv;
          if (RELU) x = fmaxf(x, 0.0f);
          if (HALF_OUT)
            ((_Float16*)Cout)[(size_t)row * Nn + col] = (_Float16)x;
          else
            ((float*)Cout)[(size_t)row * Nn + col] = x;
        }
      }
    }
  }
}

// ---------------- embedding + sinusoidal positional encoding ----------------
__global__ void embed_kernel(const int* __restrict__ x, const float* __restrict__ emb,
                             float* __restrict__ h) {
  const int t = blockIdx.x;
  const int lane = threadIdx.x;
  const int tok = x[t];
  const int s = t & (SEQ - 1);
#pragma unroll
  for (int i = 0; i < 8; ++i) {
    const int d = lane + 32 * i;
    const int de = d & ~1;
    const float freq = __expf(-(float)de * (9.210340371976184f / (float)DMODEL));
    const float ang = (float)s * freq;
    const float pe = (d & 1) ? __cosf(ang) : __sinf(ang);
    h[(size_t)t * DMODEL + d] = emb[(size_t)tok * DMODEL + d] * 16.0f + pe;
  }
}

__global__ void pad_kernel(const float* __restrict__ h, int* __restrict__ pad) {
  const int t = blockIdx.x;
  const int lane = threadIdx.x;
  float s = 0.0f;
#pragma unroll
  for (int i = 0; i < 8; ++i) s += h[(size_t)t * DMODEL + lane + 32 * i];
  for (int m = 1; m < 32; m <<= 1) s += __shfl_xor(s, m, 32);
  if (lane == 0) pad[t] = (s == 0.0f) ? 1 : 0;
}

// ---------------- V transpose: qkv[:,512+h*32+d] -> vT[(b,h,d), s] ----------------
__global__ void transpose_v(const _Float16* __restrict__ qkv, _Float16* __restrict__ vT) {
  __shared__ _Float16 tile[32][33];
  const int bh = blockIdx.x;
  const int b = bh >> 3, hh = bh & 7;
  const int st = blockIdx.y * 32;
  const int tx = threadIdx.x & 31;
  const int ty = threadIdx.x >> 5;
  for (int r = ty; r < 32; r += 8)
    tile[r][tx] = qkv[(size_t)(b * SEQ + st + r) * 768 + 512 + hh * 32 + tx];
  __syncthreads();
  for (int r = ty; r < 32; r += 8)
    vT[((size_t)bh * DHEAD + r) * SEQ + st + tx] = tile[tx][r];
}

// ---------------- flash attention: one wave per (b,h, 32-query tile) ----------------
// Softmax is shift-invariant and logits here are O(1), so we skip the running-max
// rescale (exp cannot overflow fp32; the pad mask maps directly to p=0) and defer
// the row-sum lane reduction to a single epilogue pass. This keeps persistent wave
// state to aq(16)+acc(32)+lpart(16) VGPRs -> no spills around the 8 WMMAs/iter.
__global__ void flash_attn(const _Float16* __restrict__ qkv, const _Float16* __restrict__ vT,
                           const int* __restrict__ pad, _Float16* __restrict__ ctx) {
  __shared__ _Float16 Pl[32 * 32];
  const int lane = threadIdx.x;
  const int g = lane >> 4, mr = lane & 15;
  const int bh = blockIdx.x;
  const int b = bh >> 3, hh = bh & 7;
  const int qt = blockIdx.y;                 // 32 queries per wave
  const float scale = 0.17677669529663687f;  // 1/sqrt(32)

  // two Q tiles (A matrices, 16x32 each)
  HV aq[2];
#pragma unroll
  for (int qi = 0; qi < 2; ++qi) {
    const int qs = qt * 32 + qi * 16 + mr;
    const _Float16* qp = qkv + (size_t)(b * SEQ + qs) * 768 + hh * 32;
    aq[qi].h[0] = *(const v8h*)(qp + 8 * g);
    aq[qi].h[1] = *(const v8h*)(qp + 16 + 8 * g);
  }

  CV acc[2][2];
  float lpart[2][8];                         // per-lane partial row sums
#pragma unroll
  for (int qi = 0; qi < 2; ++qi)
#pragma unroll
    for (int j = 0; j < 8; ++j) {
      acc[qi][0].f[j] = 0.0f; acc[qi][1].f[j] = 0.0f;
      lpart[qi][j] = 0.0f;
    }
  CV zc;
#pragma unroll
  for (int j = 0; j < 8; ++j) zc.f[j] = 0.0f;

  const _Float16* kbase = qkv + 256 + hh * 32;
  const _Float16* vrow0 = vT + ((size_t)bh * DHEAD + mr) * SEQ;
  const _Float16* vrow1 = vT + ((size_t)bh * DHEAD + 16 + mr) * SEQ;

  for (int kb = 0; kb < SEQ; kb += 32) {
    const int key0 = kb + mr, key1 = kb + 16 + mr;
    HV b0, b1;
    {
      const _Float16* kp0 = kbase + (size_t)(b * SEQ + key0) * 768;
      const _Float16* kp1 = kbase + (size_t)(b * SEQ + key1) * 768;
      b0.h[0] = *(const v8h*)(kp0 + 8 * g);
      b0.h[1] = *(const v8h*)(kp0 + 16 + 8 * g);
      b1.h[0] = *(const v8h*)(kp1 + 8 * g);
      b1.h[1] = *(const v8h*)(kp1 + 16 + 8 * g);
    }
    const int msk0 = pad[b * SEQ + key0];
    const int msk1 = pad[b * SEQ + key1];

#pragma unroll
    for (int qi = 0; qi < 2; ++qi) {
      CV s0, s1;
      s0.v = wmma_f16(aq[qi].v, b0.v, zc.v);
      s1.v = wmma_f16(aq[qi].v, b1.v, zc.v);
#pragma unroll
      for (int j = 0; j < 8; ++j) {
        const float p0 = msk0 ? 0.0f : __expf(s0.f[j] * scale);
        const float p1 = msk1 ? 0.0f : __expf(s1.f[j] * scale);
        lpart[qi][j] += p0 + p1;
        Pl[(qi * 16 + j + 8 * g) * 32 + mr] = (_Float16)p0;
        Pl[(qi * 16 + j + 8 * g) * 32 + 16 + mr] = (_Float16)p1;
      }
    }
    __syncthreads();

    // P (two 16x32 A tiles from LDS); V tiles (32x16 B each) shared by both
    HV ap[2], bv0, bv1;
    ap[0].h[0] = *(const v8h*)&Pl[mr * 32 + 8 * g];
    ap[0].h[1] = *(const v8h*)&Pl[mr * 32 + 16 + 8 * g];
    ap[1].h[0] = *(const v8h*)&Pl[(16 + mr) * 32 + 8 * g];
    ap[1].h[1] = *(const v8h*)&Pl[(16 + mr) * 32 + 16 + 8 * g];
    bv0.h[0] = *(const v8h*)(vrow0 + kb + 8 * g);
    bv0.h[1] = *(const v8h*)(vrow0 + kb + 16 + 8 * g);
    bv1.h[0] = *(const v8h*)(vrow1 + kb + 8 * g);
    bv1.h[1] = *(const v8h*)(vrow1 + kb + 16 + 8 * g);
#pragma unroll
    for (int qi = 0; qi < 2; ++qi) {
      acc[qi][0].v = wmma_f16(ap[qi].v, bv0.v, acc[qi][0].v);
      acc[qi][1].v = wmma_f16(ap[qi].v, bv1.v, acc[qi][1].v);
    }
    __syncthreads();
  }

  // epilogue: one 16-lane reduction of the deferred row sums, then normalize+store
#pragma unroll
  for (int qi = 0; qi < 2; ++qi) {
#pragma unroll
    for (int j = 0; j < 8; ++j) {
      float s = lpart[qi][j];
      for (int m = 1; m < 16; m <<= 1) s += __shfl_xor(s, m, 32);
      const float inv = 1.0f / s;
      const int qs = qt * 32 + qi * 16 + j + 8 * g;
      _Float16* cp = ctx + (size_t)(b * SEQ + qs) * DMODEL + hh * 32;
      cp[mr] = (_Float16)(acc[qi][0].f[j] * inv);
      cp[16 + mr] = (_Float16)(acc[qi][1].f[j] * inv);
    }
  }
}

// ---------------- layernorm(h + delta) ; delta is dense or routed-MoE gather ----------------
__global__ void ln_kernel(const float* __restrict__ h, const float* __restrict__ delta,
                          const float* __restrict__ Yg, const int* __restrict__ rev,
                          const float* __restrict__ topw, const float* __restrict__ gamma,
                          const float* __restrict__ beta, float* __restrict__ out) {
  const int t = blockIdx.x;
  const int lane = threadIdx.x;
  float x[8];
  float sw = 1.0f;
  const float* drow;
  if (Yg) { sw = topw[t]; drow = Yg + (size_t)rev[t] * DMODEL; }
  else    { drow = delta + (size_t)t * DMODEL; }

  float s = 0.0f;
#pragma unroll
  for (int i = 0; i < 8; ++i) {
    const int d = lane + 32 * i;
    x[i] = h[(size_t)t * DMODEL + d] + sw * drow[d];
    s += x[i];
  }
  for (int m = 1; m < 32; m <<= 1) s += __shfl_xor(s, m, 32);
  const float mu = s * (1.0f / DMODEL);
  float v = 0.0f;
#pragma unroll
  for (int i = 0; i < 8; ++i) { const float e = x[i] - mu; v += e * e; }
  for (int m = 1; m < 32; m <<= 1) v += __shfl_xor(v, m, 32);
  const float r = rsqrtf(v * (1.0f / DMODEL) + 1e-5f);
#pragma unroll
  for (int i = 0; i < 8; ++i) {
    const int d = lane + 32 * i;
    out[(size_t)t * DMODEL + d] = (x[i] - mu) * r * gamma[d] + beta[d];
  }
}

// ---------------- gate: softmax over 8 experts, top-1 ----------------
__global__ void gate_kernel(const float* __restrict__ h, const float* __restrict__ gw,
                            const float* __restrict__ gb, int* __restrict__ tope,
                            float* __restrict__ topw) {
  const int t = blockIdx.x * blockDim.x + threadIdx.x;
  if (t >= NTOK) return;
  const float* x = h + (size_t)t * DMODEL;
  float lg[NEXP];
#pragma unroll
  for (int e = 0; e < NEXP; ++e) {
    const float* w = gw + (size_t)e * DMODEL;
    float s = gb[e];
    for (int k = 0; k < DMODEL; ++k) s += x[k] * w[k];
    lg[e] = s;
  }
  float mx = lg[0]; int bi = 0;
#pragma unroll
  for (int e = 1; e < NEXP; ++e)
    if (lg[e] > mx) { mx = lg[e]; bi = e; }
  float den = 0.0f;
#pragma unroll
  for (int e = 0; e < NEXP; ++e) den += __expf(lg[e] - mx);
  tope[t] = bi;
  topw[t] = 1.0f / den;   // softmax value at argmax
}

__global__ void zero_int(int* __restrict__ p, int n) {
  const int i = blockIdx.x * blockDim.x + threadIdx.x;
  if (i < n) p[i] = 0;
}

__global__ void moe_count(const int* __restrict__ tope, int* __restrict__ counts) {
  const int t = blockIdx.x * blockDim.x + threadIdx.x;
  if (t < NTOK) atomicAdd(&counts[tope[t]], 1);
}

__global__ void moe_offsets(const int* __restrict__ counts, int* __restrict__ eoff) {
  if (threadIdx.x == 0 && blockIdx.x == 0) {
    int acc = 0;
    for (int e = 0; e < NEXP; ++e) { eoff[e] = acc; acc += (counts[e] + 31) & ~31; }
    eoff[NEXP] = acc;
  }
}

__global__ void moe_scatter(const int* __restrict__ tope, int* __restrict__ cursor,
                            const int* __restrict__ eoff, int* __restrict__ perm,
                            int* __restrict__ rev) {
  const int t = blockIdx.x * blockDim.x + threadIdx.x;
  if (t >= NTOK) return;
  const int e = tope[t];
  const int slot = atomicAdd(&cursor[e], 1);
  const int idx = eoff[e] + slot;
  perm[idx] = t;
  rev[t] = idx;
}

// ---------------- masked mean pool over sequence ----------------
__global__ void pool_kernel(const float* __restrict__ h, const int* __restrict__ pad,
                            float* __restrict__ pooled) {
  const int b = blockIdx.x;
  const int d = threadIdx.x;
  float s = 0.0f, cnt = 0.0f;
  for (int si = 0; si < SEQ; ++si) {
    const int t = b * SEQ + si;
    if (!pad[t]) { s += h[(size_t)t * DMODEL + d]; cnt += 1.0f; }
  }
  pooled[b * DMODEL + d] = s / fmaxf(cnt, 1.0f);
}

__global__ void f32_to_f16(const float* __restrict__ src, _Float16* __restrict__ dst, int n) {
  const int i = blockIdx.x * blockDim.x + threadIdx.x;
  if (i < n) dst[i] = (_Float16)src[i];
}

// =======================================================================
extern "C" void kernel_launch(void* const* d_in, const int* in_sizes, int n_in,
                              void* d_out, int out_size, void* d_ws, size_t ws_size,
                              hipStream_t stream) {
  (void)in_sizes; (void)n_in; (void)out_size; (void)ws_size;

  const int*   x      = (const int*)  d_in[0];
  const float* emb    = (const float*)d_in[1];
  const float* in_w   = (const float*)d_in[2];
  const float* in_b   = (const float*)d_in[3];
  const float* out_w  = (const float*)d_in[4];
  const float* out_b  = (const float*)d_in[5];
  const float* ln1_g  = (const float*)d_in[6];
  const float* ln1_b  = (const float*)d_in[7];
  const float* ln2_g  = (const float*)d_in[8];
  const float* ln2_b  = (const float*)d_in[9];
  const float* gate_w = (const float*)d_in[10];
  const float* gate_b = (const float*)d_in[11];
  const float* w1     = (const float*)d_in[12];
  const float* b1     = (const float*)d_in[13];
  const float* w2     = (const float*)d_in[14];
  const float* b2     = (const float*)d_in[15];
  const float* fc1_w  = (const float*)d_in[16];
  const float* fc1_b  = (const float*)d_in[17];
  const float* fc2_w  = (const float*)d_in[18];
  const float* fc2_b  = (const float*)d_in[19];

  char* base = (char*)d_ws;
  size_t off = 0;
  auto carve = [&](size_t bytes) -> void* {
    void* p = base + off;
    off = (off + bytes + 255) & ~(size_t)255;
    return p;
  };

  float*    h      = (float*)   carve((size_t)NTOK * DMODEL * 4);
  int*      pad    = (int*)     carve((size_t)NTOK * 4);
  _Float16* qkvh   = (_Float16*)carve((size_t)NTOK * 768 * 2);
  _Float16* vT     = (_Float16*)carve((size_t)BATCH * NHEAD * DHEAD * SEQ * 2);
  _Float16* ctxh   = (_Float16*)carve((size_t)NTOK * DMODEL * 2);
  float*    attn   = (float*)   carve((size_t)NTOK * DMODEL * 4);
  int*      tope   = (int*)     carve((size_t)NTOK * 4);
  float*    topw   = (float*)   carve((size_t)NTOK * 4);
  int*      rev    = (int*)     carve((size_t)NTOK * 4);
  int*      perm   = (int*)     carve((size_t)PERMCAP * 4);
  int*      mcnt   = (int*)     carve(16 * 4);            // counts[8] | cursor[8]
  int*      eoff   = (int*)     carve(16 * 4);            // offsets[9]
  _Float16* Hg     = (_Float16*)carve((size_t)PERMCAP * EHID * 2);
  float*    Yg     = (float*)   carve((size_t)PERMCAP * DMODEL * 4);
  float*    pooled = (float*)   carve((size_t)BATCH * DMODEL * 4);
  float*    zbuf   = (float*)   carve((size_t)BATCH * 128 * 4);
  _Float16* wqkvh  = (_Float16*)carve((size_t)2 * 768 * 256 * 2);
  _Float16* wouth  = (_Float16*)carve((size_t)2 * 256 * 256 * 2);
  _Float16* w1h    = (_Float16*)carve((size_t)2 * NEXP * EHID * 256 * 2);
  _Float16* w2h    = (_Float16*)carve((size_t)2 * NEXP * 256 * EHID * 2);
  _Float16* fc1h   = (_Float16*)carve((size_t)128 * 256 * 2);
  _Float16* fc2h   = (_Float16*)carve((size_t)2 * 128 * 2);

  int* counts = mcnt;
  int* cursor = mcnt + 8;

  // --- weight conversion to f16 (kept in [out,in] row-major == WMMA B layout) ---
  {
    int n;
    n = 2 * 768 * 256;         f32_to_f16<<<(n + 255) / 256, 256, 0, stream>>>(in_w, wqkvh, n);
    n = 2 * 256 * 256;         f32_to_f16<<<(n + 255) / 256, 256, 0, stream>>>(out_w, wouth, n);
    n = 2 * NEXP * EHID * 256; f32_to_f16<<<(n + 255) / 256, 256, 0, stream>>>(w1, w1h, n);
    n = 2 * NEXP * 256 * EHID; f32_to_f16<<<(n + 255) / 256, 256, 0, stream>>>(w2, w2h, n);
    n = 128 * 256;             f32_to_f16<<<(n + 255) / 256, 256, 0, stream>>>(fc1_w, fc1h, n);
    n = 2 * 128;               f32_to_f16<<<(n + 255) / 256, 256, 0, stream>>>(fc2_w, fc2h, n);
  }

  // --- embedding + positional encoding + pad mask ---
  embed_kernel<<<NTOK, 32, 0, stream>>>(x, emb, h);
  pad_kernel<<<NTOK, 32, 0, stream>>>(h, pad);

  for (int l = 0; l < 2; ++l) {
    // QKV projection -> f16 [N,768]
    wmma_gemm<float, false, true><<<dim3(NTOK / 32, 768 / 64), 32, 0, stream>>>(
        h, wqkvh + (size_t)l * 768 * 256, in_b + l * 768, qkvh, NTOK, 768, 256);

    // V transpose for contiguous B tiles in P@V
    transpose_v<<<dim3(BATCH * NHEAD, SEQ / 32), 256, 0, stream>>>(qkvh, vT);

    // flash attention -> ctx f16 [N,256]
    flash_attn<<<dim3(BATCH * NHEAD, SEQ / 32), 32, 0, stream>>>(qkvh, vT, pad, ctxh);

    // output projection -> attn fp32 [N,256]
    wmma_gemm<_Float16, false, false><<<dim3(NTOK / 32, 256 / 64), 32, 0, stream>>>(
        ctxh, wouth + (size_t)l * 256 * 256, out_b + l * 256, attn, NTOK, 256, 256);

    // h = LN1(h + attn)
    ln_kernel<<<NTOK, 32, 0, stream>>>(h, attn, nullptr, nullptr, nullptr,
                                       ln1_g + l * 256, ln1_b + l * 256, h);

    // --- top-1 MoE routing ---
    gate_kernel<<<NTOK / 256, 256, 0, stream>>>(h, gate_w + l * NEXP * 256,
                                                gate_b + l * NEXP, tope, topw);
    zero_int<<<1, 32, 0, stream>>>(mcnt, 16);
    zero_int<<<(PERMCAP + 255) / 256, 256, 0, stream>>>(perm, PERMCAP);
    moe_count<<<NTOK / 256, 256, 0, stream>>>(tope, counts);
    moe_offsets<<<1, 1, 0, stream>>>(counts, eoff);
    moe_scatter<<<NTOK / 256, 256, 0, stream>>>(tope, cursor, eoff, perm, rev);

    // routed expert FFN: Hg = relu(X[perm] @ w1_e^T + b1_e)  (f16 out)
    wmma_gemm_moe<float, true, true><<<dim3(PERMCAP / 32, EHID / 64), 32, 0, stream>>>(
        h, w1h + (size_t)l * NEXP * EHID * 256, b1 + l * NEXP * EHID, Hg,
        eoff, perm, EHID, 256);
    // Yg = Hg @ w2_e^T + b2_e  (fp32 out)
    wmma_gemm_moe<_Float16, false, false><<<dim3(PERMCAP / 32, 256 / 64), 32, 0, stream>>>(
        Hg, w2h + (size_t)l * NEXP * 256 * EHID, b2 + l * NEXP * 256, Yg,
        eoff, nullptr, 256, EHID);

    // h = LN2(h + top_w * Yg[rev])
    ln_kernel<<<NTOK, 32, 0, stream>>>(h, nullptr, Yg, rev, topw,
                                       ln2_g + l * 256, ln2_b + l * 256, h);
  }

  // --- masked mean pool + classifier head ---
  pool_kernel<<<BATCH, DMODEL, 0, stream>>>(h, pad, pooled);
  wmma_gemm<float, true, false><<<dim3(1, 2), 32, 0, stream>>>(
      pooled, fc1h, fc1_b, zbuf, BATCH, 128, 256);
  wmma_gemm<float, false, false><<<dim3(1, 1), 32, 0, stream>>>(
      zbuf, fc2h, fc2_b, (float*)d_out, BATCH, 2, 128);
}
